// ICTDO3E3Conv_84344567759197
// MI455X (gfx1250) — matrix operand order repaired and checked by reading, not compiled, not run on hardware
//
#include <hip/hip_runtime.h>
#include <hip/hip_bf16.h>

typedef __attribute__((ext_vector_type(16))) _Float16 v16h;
typedef __attribute__((ext_vector_type(8)))  _Float16 v8h;
typedef __attribute__((ext_vector_type(8)))  float    v8f;

#define WMMA(a,b,c) __builtin_amdgcn_wmma_f32_16x16x32_f16(false,(a),false,(b),(short)0,(c),false,false)

// ---- canonical path metadata ----------------------------------------------
// PATHS enumeration order of the reference; CANON = indices with p==(-1)^l.
__constant__ int d_CANON[16] = {0,3,4,24,29,30,31,32,33,36,40,41,42,43,44,0};
__constant__ int d_PL1[15]   = {0,0,0,1,1,1,1,1,1,2,2,2,2,2,2};
__constant__ int d_PL2[15]   = {0,1,2,0,1,1,1,2,2,0,1,1,2,2,2};
__constant__ int d_PLO[15]   = {0,1,2,1,0,1,2,1,2,2,1,2,0,1,2};
__constant__ int d_CGOFF[15] = {0,1,10,35,44,53,80,125,170,245,270,315,390,415,490};
// total CG floats = 615

// fast silu: v_exp_f32 + v_rcp_f32 (avoids IEEE div expansion)
__device__ __forceinline__ float siluf(float x) {
  return x * __builtin_amdgcn_rcpf(1.0f + __expf(-x));
}

// ---- Clebsch-Gordan setup kernel (real basis), 15 blocks, one per path ----
__device__ inline void qent(int l, int a, int i, double& re, double& im) {
  double r = 0.0, m = 0.0;
  const double inv = 0.70710678118654752440;
  int mm = a - l;
  if (mm < 0) {
    if (i == 2*l - a) r = inv;
    if (i == a)       m = -inv;
  } else if (mm == 0) {
    if (i == l) r = 1.0;
  } else {
    double sgn = (mm & 1) ? -1.0 : 1.0;
    if (i == a)       r = sgn * inv;
    if (i == 2*l - a) m = sgn * inv;
  }
  if (l == 1) { double t = r; r = m; m = -t; }   // multiply by (-i)
  else if (l == 2) { r = -r; m = -m; }           // multiply by (-1)
  re = r; im = m;
}

__global__ void cg_kernel(float* __restrict__ cgout) {
  __shared__ double su2[125];
  const int p  = blockIdx.x;
  const int l1 = d_PL1[p], l2 = d_PL2[p], l3 = d_PLO[p];
  const int n1 = 2*l1+1, n2 = 2*l2+1, n3 = 2*l3+1;
  const int tot = n1*n2*n3;
  const double F[8] = {1,1,2,6,24,120,720,5040};
  for (int idx = threadIdx.x; idx < tot; idx += blockDim.x) {
    int a = idx/(n2*n3), b = (idx/n3)%n2, c = idx%n3;
    int m1 = a-l1, m2 = b-l2, m3 = c-l3;
    double val = 0.0;
    if (m1+m2 == m3) {
      double pref = sqrt((double)(2*l3+1)*F[l1+l2-l3]*F[l3+l1-l2]*F[l3+l2-l1]/F[l1+l2+l3+1]);
      pref *= sqrt(F[l3+m3]*F[l3-m3]*F[l1-m1]*F[l1+m1]*F[l2-m2]*F[l2+m2]);
      double s = 0.0;
      for (int v = 0; v <= l1+l2-l3; v++) {
        int t1=l1+l2-l3-v, t2=l1-m1-v, t3=l2+m2-v, t4=l3-l2+m1+v, t5=l3-l1-m2+v;
        if (t1<0||t2<0||t3<0||t4<0||t5<0) continue;
        s += ((v&1)?-1.0:1.0)/(F[v]*F[t1]*F[t2]*F[t3]*F[t4]*F[t5]);
      }
      val = pref * s;
    }
    su2[idx] = val;
  }
  __syncthreads();
  for (int idx = threadIdx.x; idx < tot; idx += blockDim.x) {
    int i = idx/(n2*n3), j = (idx/n3)%n2, k = idx%n3;
    double acc = 0.0;
    for (int a = 0; a < n1; a++) {
      double r1, i1; qent(l1, a, i, r1, i1);
      if (r1 == 0.0 && i1 == 0.0) continue;
      for (int b = 0; b < n2; b++) {
        int c = a + b - l1 - l2 + l3;
        if (c < 0 || c >= n3) continue;
        double s = su2[(a*n2+b)*n3 + c];
        if (s == 0.0) continue;
        double r2, i2; qent(l2, b, j, r2, i2);
        double r3, i3; qent(l3, c, k, r3, i3);
        double rr = r1*r2 - i1*i2, ii = r1*i2 + i1*r2;     // Q1*Q2
        acc += s * (rr*r3 + ii*i3);                        // * conj(Q3), real part
      }
    }
    cgout[d_CGOFF[p] + idx] = (float)acc;
  }
}

// ---- zero the output -------------------------------------------------------
__global__ void zero_kernel(float* __restrict__ out, int n) {
  int i = blockIdx.x*blockDim.x + threadIdx.x;
  int stride = gridDim.x*blockDim.x;
  for (; i < n; i += stride) out[i] = 0.0f;
}

// ---- node MLP: Ai = silu(emb[A]@aw1+ab1)@aw2+ab2 ---------------------------
__global__ __launch_bounds__(256) void node_kernel(
    const float* __restrict__ emb, const float* __restrict__ aw1,
    const float* __restrict__ ab1, const float* __restrict__ aw2,
    const float* __restrict__ ab2, const int* __restrict__ A,
    float* __restrict__ Ai, int N) {
  __shared__ float s_emb[100*16];
  __shared__ float s_w1[16*64];
  __shared__ float s_w2[64*8];
  __shared__ float s_b1[64];
  __shared__ float s_b2[8];
  const int tid = threadIdx.x;
  for (int i = tid; i < 1600; i += 256) s_emb[i] = emb[i];
  for (int i = tid; i < 1024; i += 256) s_w1[i] = aw1[i];
  for (int i = tid; i < 512;  i += 256) s_w2[i] = aw2[i];
  if (tid < 64) s_b1[tid] = ab1[tid];
  if (tid < 8)  s_b2[tid] = ab2[tid];
  __syncthreads();
  int n = blockIdx.x*256 + tid;
  if (n >= N) return;
  int a = A[n];
  float x[16];
  #pragma unroll
  for (int i = 0; i < 16; i++) x[i] = s_emb[a*16 + i];
  float o[8];
  #pragma unroll
  for (int u = 0; u < 8; u++) o[u] = s_b2[u];
  for (int j = 0; j < 64; j++) {
    float h = s_b1[j];
    #pragma unroll
    for (int i = 0; i < 16; i++) h += x[i]*s_w1[i*64 + j];
    h = siluf(h);
    #pragma unroll
    for (int u = 0; u < 8; u++) o[u] += h*s_w2[j*8 + u];
  }
  #pragma unroll
  for (int u = 0; u < 8; u++) Ai[n*8 + u] = o[u];
}

// ---- WMMA fragment loaders -------------------------------------------------
// A (16x32 f16): lane<16 holds row m=lane, K = kb+h; kb=16 for hi lanes
__device__ __forceinline__ v16h load_a16(const _Float16* row, int kb) {
  v8h lo = *(const v8h*)(row + kb);
  v8h hi = *(const v8h*)(row + kb + 8);
  return __builtin_shufflevector(lo, hi, 0,1,2,3,4,5,6,7,8,9,10,11,12,13,14,15);
}

// ---- TP group: wmma + register accumulate + atomic scatter -----------------
template<int NP, int K>
__device__ __forceinline__ void tp_group(
    const int (&paths)[NP], int OFF,
    v16h sa0, v16h sa1,
    const _Float16* __restrict__ wtp,   // [240][64] B^T
    const float* __restrict__ pwc,      // [16][16] (pw * alpha)
    const float* __restrict__ gbuf,     // [16][75]
    const int* __restrict__ dstb,       // [16]
    float* __restrict__ out,
    int hl, int isHi, int tileBase, int E) {
  float acc[8][K];
  #pragma unroll
  for (int r = 0; r < 8; r++)
    #pragma unroll
    for (int k = 0; k < K; k++) acc[r][k] = 0.0f;
  #pragma unroll
  for (int pi = 0; pi < NP; pi++) {
    const int p = paths[pi];
    const _Float16* brow = wtp + (p*16 + hl)*64 + isHi*16;
    v16h b0 = *(const v16h*)(brow);
    v16h b1 = *(const v16h*)(brow + 32);
    v8f d = {0.f,0.f,0.f,0.f,0.f,0.f,0.f,0.f};
    d = WMMA(sa0, b0, d);
    d = WMMA(sa1, b1, d);
    #pragma unroll
    for (int r = 0; r < 8; r++) {
      int m = r + isHi*8;
      float t = d[r] * pwc[m*16 + p];
      #pragma unroll
      for (int k = 0; k < K; k++) acc[r][k] += t * gbuf[m*75 + p*5 + k];
    }
  }
  #pragma unroll
  for (int r = 0; r < 8; r++) {
    int m = r + isHi*8;
    if (tileBase + m < E) {
      float* o = out + (size_t)dstb[m]*288 + OFF + hl*K;
      #pragma unroll
      for (int k = 0; k < K; k++) unsafeAtomicAdd(o + k, acc[r][k]);
    }
  }
}

#define TILES_PER_BLOCK 4

// ---- fused edge kernel -----------------------------------------------------
__global__ __launch_bounds__(256) void edge_kernel(
    const float* __restrict__ pos, const int* __restrict__ batch,
    const int* __restrict__ esrc, const int* __restrict__ edst,
    const float* __restrict__ eshift, const float* __restrict__ cell,
    const float* __restrict__ fw1, const float* __restrict__ fb1,
    const float* __restrict__ fw2, const float* __restrict__ fb2,
    const float* __restrict__ fw3, const float* __restrict__ fb3,
    const float* __restrict__ tpw, const float* __restrict__ cgw,
    const float* __restrict__ Ai, float* __restrict__ out, int E) {
  __shared__ alignas(32) _Float16 s_fw1t[64*32];    // [n][k], K padded 8->32
  __shared__ alignas(32) _Float16 s_fw2t[64*64];    // [n][k]
  __shared__ alignas(32) _Float16 s_fw3t[16*64];    // [n(canon path)][k]
  __shared__ alignas(32) _Float16 s_wtpT[240*64];   // [p*16+w][u*8+v]
  __shared__ float s_fb1[64], s_fb2[64], s_fb3[16];
  __shared__ float s_cg[615];
  __shared__ alignas(16) _Float16 s_hbuf[8][16*64]; // per-wave [edge][hid]
  __shared__ float s_ybuf[8][16*8];                 // per-wave Y1(3)+Y2(5)
  __shared__ float s_pwc[8][16*16];                 // per-wave pw*alpha
  __shared__ float s_gbuf[8][16*75];                // per-wave G[e][p][k]
  __shared__ int   s_dstn[8][16];

  const int tid = threadIdx.x;
  // ---- stage weights (transposed, f16) ----
  for (int idx = tid; idx < 64*32; idx += 256) {
    int n = idx >> 5, k = idx & 31;
    s_fw1t[idx] = (k < 8) ? (_Float16)fw1[k*64 + n] : (_Float16)0.0f;
  }
  for (int idx = tid; idx < 64*64; idx += 256) {
    int n = idx >> 6, k = idx & 63;
    s_fw2t[idx] = (_Float16)fw2[k*64 + n];
  }
  for (int idx = tid; idx < 16*64; idx += 256) {
    int n = idx >> 6, k = idx & 63;
    s_fw3t[idx] = (n < 15) ? (_Float16)fw3[k*60 + d_CANON[n]] : (_Float16)0.0f;
  }
  for (int idx = tid; idx < 240*64; idx += 256) {
    int col = idx >> 6, k = idx & 63;
    int p = col >> 4, w = col & 15, u = k >> 3, v = k & 7;
    s_wtpT[idx] = (_Float16)tpw[((d_CANON[p]*8 + u)*8 + v)*16 + w];
  }
  if (tid < 64) { s_fb1[tid] = fb1[tid]; s_fb2[tid] = fb2[tid]; }
  if (tid < 16) s_fb3[tid] = (tid < 15) ? fb3[d_CANON[tid]] : 0.0f;
  for (int idx = tid; idx < 615; idx += 256) s_cg[idx] = cgw[idx];
  __syncthreads();

  const int wave = tid >> 5, lane = tid & 31, hl = lane & 15, isHi = lane >> 4;
  _Float16* hbuf = &s_hbuf[wave][0];
  float* ybuf = &s_ybuf[wave][0];
  float* pwc  = &s_pwc[wave][0];
  float* gbuf = &s_gbuf[wave][0];
  int*   dstb = &s_dstn[wave][0];

  for (int it = 0; it < TILES_PER_BLOCK; it++) {
    const int tileBase = (blockIdx.x*(8*TILES_PER_BLOCK) + it*8 + wave)*16;
    int e = tileBase + hl; if (e >= E) e = E - 1;

    // ---- per-edge scalars (both half-waves redundantly) ----
    const int vS = esrc[e], vD = edst[e];
    const int bg = batch[vS];
    const float* cl = cell + bg*9;
    const float sh0 = eshift[e*3+0], sh1 = eshift[e*3+1], sh2 = eshift[e*3+2];
    const float shx = sh0*cl[0] + sh1*cl[3] + sh2*cl[6];
    const float shy = sh0*cl[1] + sh1*cl[4] + sh2*cl[7];
    const float shz = sh0*cl[2] + sh1*cl[5] + sh2*cl[8];
    const float ex = pos[vD*3+0] - pos[vS*3+0] + shx;
    const float ey = pos[vD*3+1] - pos[vS*3+1] + shy;
    const float ez = pos[vD*3+2] - pos[vS*3+2] + shz;
    const float el = sqrtf(ex*ex + ey*ey + ez*ez);
    const float rn = __builtin_amdgcn_rcpf(fmaxf(el, 1e-8f));
    const float nx = ex*rn, ny = ey*rn, nz = ez*rn;
    const float S3 = 1.7320508075688772f, S5 = 2.2360679774997896f, S15 = 3.8729833462074170f;
    float Yv[8];
    Yv[0] = S3*nx; Yv[1] = S3*ny; Yv[2] = S3*nz;
    Yv[3] = S15*nx*ny; Yv[4] = S15*ny*nz; Yv[5] = 0.5f*S5*(3.0f*nz*nz - 1.0f);
    Yv[6] = S15*nx*nz; Yv[7] = 0.5f*S15*(nx*nx - ny*ny);
    if (!isHi) {
      #pragma unroll
      for (int i = 0; i < 8; i++) ybuf[hl*8 + i] = Yv[i];
      dstb[hl] = vD;
    }
    float asrc[8], adst[8];
    #pragma unroll
    for (int u = 0; u < 8; u++) { asrc[u] = Ai[vS*8 + u]; adst[u] = Ai[vD*8 + u]; }
    float rbv[8];
    #pragma unroll
    for (int k = 0; k < 8; k++) {
      float t = (el - (float)k*(5.0f/7.0f))*(7.0f/5.0f);
      rbv[k] = __expf(-t*t);
    }
    __syncthreads();

    // ---- radial MLP layer 1: [16x8(pad32)] @ [32x64] ----
    v16h a0;
    #pragma unroll
    for (int h = 0; h < 16; h++)
      a0[h] = (_Float16)((h < 8 && !isHi) ? rbv[h & 7] : 0.0f);
    #pragma unroll
    for (int nc = 0; nc < 4; nc++) {
      const int n = nc*16 + hl;
      v16h b = *(const v16h*)(s_fw1t + n*32 + isHi*16);
      float bias = s_fb1[n];
      v8f c = {bias,bias,bias,bias,bias,bias,bias,bias};
      c = WMMA(a0, b, c);
      #pragma unroll
      for (int r = 0; r < 8; r++) hbuf[(r + isHi*8)*64 + n] = (_Float16)siluf(c[r]);
    }
    __syncthreads();

    // ---- layer 2: [16x64] @ [64x64] ----
    {
      const _Float16* hrow = hbuf + hl*64;
      v16h a20 = load_a16(hrow, isHi*16);
      v16h a21 = load_a16(hrow, 32 + isHi*16);
      __syncthreads();
      #pragma unroll
      for (int nc = 0; nc < 4; nc++) {
        const int n = nc*16 + hl;
        v16h b0 = *(const v16h*)(s_fw2t + n*64 + isHi*16);
        v16h b1 = *(const v16h*)(s_fw2t + n*64 + 32 + isHi*16);
        float bias = s_fb2[n];
        v8f c = {bias,bias,bias,bias,bias,bias,bias,bias};
        c = WMMA(a20, b0, c);
        c = WMMA(a21, b1, c);
        #pragma unroll
        for (int r = 0; r < 8; r++) hbuf[(r + isHi*8)*64 + n] = (_Float16)siluf(c[r]);
      }
    }
    __syncthreads();

    // ---- layer 3 (path weights, canon cols only): [16x64] @ [64x16] ----
    {
      const _Float16* hrow = hbuf + hl*64;
      v16h a30 = load_a16(hrow, isHi*16);
      v16h a31 = load_a16(hrow, 32 + isHi*16);
      const int n = hl;
      v16h b0 = *(const v16h*)(s_fw3t + n*64 + isHi*16);
      v16h b1 = *(const v16h*)(s_fw3t + n*64 + 32 + isHi*16);
      float bias = s_fb3[n];
      v8f c = {bias,bias,bias,bias,bias,bias,bias,bias};
      c = WMMA(a30, b0, c);
      c = WMMA(a31, b1, c);
      #pragma unroll
      for (int r = 0; r < 8; r++) pwc[(r + isHi*8)*16 + n] = 0.125f * c[r];  // alpha = 1/8
    }

    // ---- G[e][p][k] = sum_ij cg[i,j,k] Y_{l1,i} Y_{l2,j} ----
    for (int t = lane; t < 240; t += 32) {
      int ei = t / 15, p = t % 15;
      int l1 = d_PL1[p], l2 = d_PL2[p], lo = d_PLO[p];
      int n1 = 2*l1 + 1, n2 = 2*l2 + 1, no = 2*lo + 1;
      const float* cg = s_cg + d_CGOFF[p];
      float g0=0.f, g1=0.f, g2=0.f, g3=0.f, g4=0.f;
      for (int i = 0; i < n1; i++) {
        float y1 = (l1 == 0) ? 1.0f : ybuf[ei*8 + ((l1 == 1) ? 0 : 3) + i];
        for (int j = 0; j < n2; j++) {
          float y2 = (l2 == 0) ? 1.0f : ybuf[ei*8 + ((l2 == 1) ? 0 : 3) + j];
          float yy = y1*y2;
          const float* cgij = cg + (i*n2 + j)*no;
          g0 += cgij[0]*yy;
          if (no > 1) { g1 += cgij[1]*yy; g2 += cgij[2]*yy; }
          if (no > 3) { g3 += cgij[3]*yy; g4 += cgij[4]*yy; }
        }
      }
      float* gb = gbuf + ei*75 + p*5;
      gb[0] = g0;
      gb[1] = (no > 1) ? g1 : 0.f; gb[2] = (no > 1) ? g2 : 0.f;
      gb[3] = (no > 3) ? g3 : 0.f; gb[4] = (no > 3) ? g4 : 0.f;
    }
    __syncthreads();

    // ---- S = src (x) dst A-fragments, K = u*8+v ----
    v16h sa0, sa1;
    #pragma unroll
    for (int h = 0; h < 16; h++) {
      const int c0 = (h < 8) ? 0 : 1;
      float au  = isHi ? asrc[c0+2] : asrc[c0];
      float au4 = isHi ? asrc[c0+6] : asrc[c0+4];
      float av  = adst[h & 7];
      sa0[h] = (_Float16)(au  * av);
      sa1[h] = (_Float16)(au4 * av);
    }

    // ---- TP GEMM per output irrep group + atomic scatter ----
    { const int gp[3] = {0,4,12};    tp_group<3,1>(gp, 0,   sa0, sa1, s_wtpT, pwc, gbuf, dstb, out, hl, isHi, tileBase, E); }
    { const int gp[2] = {5,13};      tp_group<2,3>(gp, 32,  sa0, sa1, s_wtpT, pwc, gbuf, dstb, out, hl, isHi, tileBase, E); }
    { const int gp[4] = {1,3,7,10};  tp_group<4,3>(gp, 80,  sa0, sa1, s_wtpT, pwc, gbuf, dstb, out, hl, isHi, tileBase, E); }
    { const int gp[4] = {2,6,9,14};  tp_group<4,5>(gp, 128, sa0, sa1, s_wtpT, pwc, gbuf, dstb, out, hl, isHi, tileBase, E); }
    { const int gp[2] = {8,11};      tp_group<2,5>(gp, 208, sa0, sa1, s_wtpT, pwc, gbuf, dstb, out, hl, isHi, tileBase, E); }
    __syncthreads();
  }
}

extern "C" void kernel_launch(void* const* d_in, const int* in_sizes, int n_in,
                              void* d_out, int out_size, void* d_ws, size_t ws_size,
                              hipStream_t stream) {
  (void)n_in; (void)ws_size;
  const float* pos    = (const float*)d_in[0];
  const int*   A      = (const int*)  d_in[1];
  const int*   batch  = (const int*)  d_in[2];
  const int*   esrc   = (const int*)  d_in[3];
  const int*   edst   = (const int*)  d_in[4];
  const float* eshift = (const float*)d_in[5];
  const float* cell   = (const float*)d_in[6];
  const float* emb    = (const float*)d_in[7];
  const float* aw1    = (const float*)d_in[8];
  const float* ab1    = (const float*)d_in[9];
  const float* aw2    = (const float*)d_in[10];
  const float* ab2    = (const float*)d_in[11];
  const float* fw1    = (const float*)d_in[12];
  const float* fb1    = (const float*)d_in[13];
  const float* fw2    = (const float*)d_in[14];
  const float* fb2    = (const float*)d_in[15];
  const float* fw3    = (const float*)d_in[16];
  const float* fb3    = (const float*)d_in[17];
  const float* tpw    = (const float*)d_in[18];
  const int N = in_sizes[1];
  const int E = in_sizes[3];
  float* out  = (float*)d_out;
  float* cgws = (float*)d_ws;          // 615 floats (reserve 1024)
  float* Ai   = (float*)d_ws + 1024;   // N*8 floats

  cg_kernel<<<15, 128, 0, stream>>>(cgws);

  int zb = (out_size + 255)/256; if (zb > 8192) zb = 8192;
  zero_kernel<<<zb, 256, 0, stream>>>(out, out_size);

  node_kernel<<<(N + 255)/256, 256, 0, stream>>>(emb, aw1, ab1, aw2, ab2, A, Ai, N);

  const int edges_per_block = 128*TILES_PER_BLOCK;
  edge_kernel<<<(E + edges_per_block - 1)/edges_per_block, 256, 0, stream>>>(
      pos, batch, esrc, edst, eshift, cell,
      fw1, fb1, fw2, fb2, fw3, fb3, tpw, cgws, Ai, out, E);
}
